// AttentionLayer_72945724555863
// MI455X (gfx1250) — matrix operand order repaired and checked by reading, not compiled, not run on hardware
//
#include <hip/hip_runtime.h>
#include <hip/hip_bf16.h>
#include <math.h>

#define B_  64
#define S_  4096
#define E_  512
#define D_  512
#define H_  512
#define NEGV (-1.0e9f)

typedef __attribute__((ext_vector_type(16))) __bf16        v16bf;
typedef __attribute__((ext_vector_type(8)))  float         v8f;
typedef __attribute__((ext_vector_type(4)))  unsigned int  v4u;

union AFrag { v4u q[2]; v16bf v; };

__device__ __forceinline__ unsigned bf16_rne(float f) {
    unsigned u = __float_as_uint(f);
    return (u + 0x7FFFu + ((u >> 16) & 1u)) >> 16;
}
__device__ __forceinline__ unsigned pack_bf16x2(float a, float b) {
    return bf16_rne(a) | (bf16_rne(b) << 16);
}

// ---------------------------------------------------------------------------
// W_enc (fp32 [E,H]) -> bf16 B-fragment layout:
//   frag[((n*16 + kc)*32 + lane)*16 + i] = W_enc[kc*32 + (lane>>4)*16 + i][n*16 + (lane&15)]
// Each lane's 16 bf16 (8 VGPRs of WMMA-B) load contiguously (32B).
// ---------------------------------------------------------------------------
__global__ void prep_wenc(const float* __restrict__ W_enc,
                          unsigned short* __restrict__ wfrag) {
    int idx  = blockIdx.x * 256 + threadIdx.x;   // 0 .. 262143
    int i    = idx & 15;
    int lane = (idx >> 4) & 31;
    int kc   = (idx >> 9) & 15;
    int n    = idx >> 13;
    int ncol = n * 16 + (lane & 15);
    int k    = kc * 32 + (lane >> 4) * 16 + i;
    wfrag[idx] = (unsigned short)bf16_rne(W_enc[k * H_ + ncol]);
}

// ---------------------------------------------------------------------------
// decoded[b][h] = sum_d dec[b,d] * W_dec[d,h]   (tiny)
// ---------------------------------------------------------------------------
__global__ void prep_decoded(const float* __restrict__ dec,
                             const float* __restrict__ W_dec,
                             float* __restrict__ decoded) {
    int idx = blockIdx.x * 256 + threadIdx.x;    // 0 .. 32767
    int b = idx >> 9, h = idx & 511;
    float s = 0.f;
    for (int d = 0; d < D_; ++d)
        s = fmaf(dec[b * D_ + d], W_dec[d * H_ + h], s);
    decoded[idx] = s;
}

// ---------------------------------------------------------------------------
// Main fused kernel. WG = 256 threads (8 waves) = 32 rows (2 M-subtiles).
// Wave w owns N-tiles n0..n0+3 (n0 = w*4).  Loop order: K outer, N inner,
// so per K-step: one A-fragment pair (ds_load_b128 x4) is reused by 8
// back-to-back WMMAs whose B fragments were issued as a single load clause.
// ---------------------------------------------------------------------------
__global__ __launch_bounds__(256) void logits_kernel(
        const float* __restrict__ enc,
        const unsigned short* __restrict__ wfrag,
        const float* __restrict__ decoded,
        const float* __restrict__ W_out,
        const int* __restrict__ inp_mask,
        float* __restrict__ logits) {
    // 32 rows x (512 + 8 pad) bf16 -> row stride 1040B (16B aligned, bank-spread)
    __shared__ __align__(16) unsigned short tile[32 * 520];
    __shared__ float red[8][32];

    const int t   = threadIdx.x;
    const int blk = blockIdx.x;          // 0 .. 8191
    const int b   = blk >> 7;            // 128 row-blocks per batch
    const int s0  = (blk & 127) << 5;    // 32 rows per block

    // Stage enc tile (fp32 -> bf16) into LDS.
    const float4* encp = (const float4*)(enc + ((size_t)(b * S_ + s0)) * E_);
#pragma unroll 4
    for (int i = 0; i < 16; ++i) {
        int idx = i * 256 + t;           // row = idx>>7, col4 = idx&127
        float4 v = encp[idx];
        unsigned lo = pack_bf16x2(v.x, v.y);
        unsigned hi = pack_bf16x2(v.z, v.w);
        *(uint2*)&tile[(idx >> 7) * 520 + (idx & 127) * 4] = make_uint2(lo, hi);
    }
    __syncthreads();

    const int w = t >> 5, lane = t & 31;
    const int m = lane & 15, hf = lane >> 4;
    const int n0 = w * 4;

    v8f acc[8];                          // [ni][mt] -> acc[ni*2+mt]
#pragma unroll
    for (int i = 0; i < 8; ++i) acc[i] = (v8f){};

    const __bf16* wbase = (const __bf16*)wfrag + (((n0 * 16) * 32 + lane) << 4);
    const unsigned short* abase0 = &tile[m * 520        + hf * 16];
    const unsigned short* abase1 = &tile[(m + 16) * 520 + hf * 16];

#pragma unroll 1
    for (int kc = 0; kc < 16; ++kc) {
        // Issue all 4 B fragments (8x global_load_b128) as one clause.
        v16bf bf[4];
#pragma unroll
        for (int ni = 0; ni < 4; ++ni)
            bf[ni] = *(const v16bf*)(wbase + ((ni * 16 + kc) << 9)); // (..)*32*16 bf16

        // A fragments for both M-subtiles: 4x ds_load_b128 (short latency).
        AFrag A0, A1;
        const unsigned short* ap0 = abase0 + kc * 32;
        const unsigned short* ap1 = abase1 + kc * 32;
        A0.q[0] = *(const v4u*)ap0;  A0.q[1] = *(const v4u*)(ap0 + 8);
        A1.q[0] = *(const v4u*)ap1;  A1.q[1] = *(const v4u*)(ap1 + 8);

#pragma unroll
        for (int ni = 0; ni < 4; ++ni) {
            acc[ni * 2]     = __builtin_amdgcn_wmma_f32_16x16x32_bf16(
                false, A0.v, false, bf[ni], (short)0, acc[ni * 2],     false, false);
            acc[ni * 2 + 1] = __builtin_amdgcn_wmma_f32_16x16x32_bf16(
                false, A1.v, false, bf[ni], (short)0, acc[ni * 2 + 1], false, false);
        }
    }

    // Fused epilogue: tanh(acc + decoded[b,ncol]) * W_out[ncol], per row.
    float lrow[2][8] = {{0.f,0.f,0.f,0.f,0.f,0.f,0.f,0.f},
                        {0.f,0.f,0.f,0.f,0.f,0.f,0.f,0.f}};
#pragma unroll
    for (int ni = 0; ni < 4; ++ni) {
        const int   ncol = (n0 + ni) * 16 + m;
        const float dv   = decoded[b * H_ + ncol];
        const float wo   = W_out[ncol];
#pragma unroll
        for (int r = 0; r < 8; ++r) {
            lrow[0][r] = fmaf(tanhf(acc[ni * 2][r]     + dv), wo, lrow[0][r]);
            lrow[1][r] = fmaf(tanhf(acc[ni * 2 + 1][r] + dv), wo, lrow[1][r]);
        }
    }

    // Reduce across the 16 N-columns held by each half-wave (rows: hf*8 + r).
#pragma unroll
    for (int mt = 0; mt < 2; ++mt)
#pragma unroll
        for (int r = 0; r < 8; ++r) {
            float v = lrow[mt][r];
            v += __shfl_xor(v, 1, 32);
            v += __shfl_xor(v, 2, 32);
            v += __shfl_xor(v, 4, 32);
            v += __shfl_xor(v, 8, 32);
            lrow[mt][r] = v;
        }
    if (m == 0) {
#pragma unroll
        for (int mt = 0; mt < 2; ++mt)
#pragma unroll
            for (int r = 0; r < 8; ++r)
                red[w][mt * 16 + hf * 8 + r] = lrow[mt][r];
    }
    __syncthreads();

    if (t < 32) {
        float v = 0.f;
#pragma unroll
        for (int ww = 0; ww < 8; ++ww) v += red[ww][t];
        int gs = b * S_ + s0 + t;
        logits[gs] = (inp_mask[gs] == 0) ? NEGV : v;
    }
}

// ---------------------------------------------------------------------------
// Row softmax over S=4096, one workgroup per batch row.
// ---------------------------------------------------------------------------
__global__ __launch_bounds__(256) void softmax_kernel(
        const float* __restrict__ logits, float* __restrict__ probs) {
    __shared__ float sd[256];
    const int b = blockIdx.x, t = threadIdx.x;
    const float* row = logits + b * S_;

    float mx = -3.4e38f;
    for (int i = t; i < S_; i += 256) mx = fmaxf(mx, row[i]);
    sd[t] = mx; __syncthreads();
    for (int o = 128; o > 0; o >>= 1) {
        if (t < o) sd[t] = fmaxf(sd[t], sd[t + o]);
        __syncthreads();
    }
    mx = sd[0]; __syncthreads();

    float sum = 0.f;
    for (int i = t; i < S_; i += 256) sum += expf(row[i] - mx);
    sd[t] = sum; __syncthreads();
    for (int o = 128; o > 0; o >>= 1) {
        if (t < o) sd[t] += sd[t + o];
        __syncthreads();
    }
    const float inv = 1.0f / sd[0];
    for (int i = t; i < S_; i += 256)
        probs[b * S_ + i] = expf(row[i] - mx) * inv;
}

// ---------------------------------------------------------------------------
// attn[b,e] = sum_s enc[b,s,e] * probs[b,s]  — streaming pass #2 over enc.
// Deterministic two-stage reduction over 8 S-chunks (no float atomics).
// ---------------------------------------------------------------------------
__global__ __launch_bounds__(256) void attn_partial(
        const float* __restrict__ enc, const float* __restrict__ probs,
        float* __restrict__ part) {
    __shared__ float sp[512];
    const int b = blockIdx.x >> 3;
    const int chunk = blockIdx.x & 7;
    const int t = threadIdx.x;
    const int sbase = chunk * 512;

    sp[t]       = probs[b * S_ + sbase + t];
    sp[t + 256] = probs[b * S_ + sbase + t + 256];
    __syncthreads();

    const float* ep = enc + ((size_t)b * S_ + sbase) * E_ + t * 2;
    float2 acc = make_float2(0.f, 0.f);
    for (int s = 0; s < 512; ++s) {
        float p = sp[s];
        float2 ev = *(const float2*)(ep + (size_t)s * E_);
        acc.x = fmaf(p, ev.x, acc.x);
        acc.y = fmaf(p, ev.y, acc.y);
    }
    float2* dst = (float2*)(part + ((size_t)(b * 8 + chunk)) * E_) + t;
    *dst = acc;
}

__global__ void attn_reduce(const float* __restrict__ part,
                            float* __restrict__ attn) {
    int idx = blockIdx.x * 256 + threadIdx.x;   // 0 .. 32767
    int b = idx >> 9, e = idx & 511;
    float s = 0.f;
#pragma unroll
    for (int c = 0; c < 8; ++c) s += part[((size_t)(b * 8 + c)) * E_ + e];
    attn[idx] = s;
}

// ---------------------------------------------------------------------------
extern "C" void kernel_launch(void* const* d_in, const int* in_sizes, int n_in,
                              void* d_out, int out_size, void* d_ws, size_t ws_size,
                              hipStream_t stream) {
    const float* enc   = (const float*)d_in[0];
    const float* dec   = (const float*)d_in[1];
    const int*   mask  = (const int*)  d_in[2];
    const float* W_enc = (const float*)d_in[3];
    const float* W_dec = (const float*)d_in[4];
    const float* W_out = (const float*)d_in[5];

    float* attn  = (float*)d_out;             // [B,E]  = 32768 floats
    float* probs = (float*)d_out + B_ * E_;   // [B,S]  = 262144 floats

    char* ws = (char*)d_ws;
    unsigned short* wfrag   = (unsigned short*)(ws);                 // 512 KB
    float*          decoded = (float*)(ws + (512u << 10));           // 128 KB
    float*          logits  = (float*)(ws + (640u << 10));           // 1 MB
    float*          part    = (float*)(ws + (1664u << 10));          // 1 MB

    prep_wenc    <<<1024, 256, 0, stream>>>(W_enc, wfrag);
    prep_decoded <<< 128, 256, 0, stream>>>(dec, W_dec, decoded);
    logits_kernel<<<8192, 256, 0, stream>>>(enc, wfrag, decoded, W_out, mask, logits);
    softmax_kernel<<< 64, 256, 0, stream>>>(logits, probs);
    attn_partial <<< 512, 256, 0, stream>>>(enc, probs, part);
    attn_reduce  <<< 128, 256, 0, stream>>>(part, attn);
}
